// LogNeuralCDE_10204842295686
// MI455X (gfx1250) — matrix operand order repaired and checked by reading, not compiled, not run on hardware
//
#include <hip/hip_runtime.h>

// Log-Neural-CDE Heun integrator for MI455X (gfx1250).
// One workgroup (8 wave32) per batch element; all weights f16-swizzled into
// LDS in v_wmma_f32_16x16x32_f16 A-fragment order; primal (column 0), then 5 tangent columns
// (columns 0-4) in a second WMMA B-tile pass (N=16). Bias folded into the WMMA
// C operand; all epilogue LDS traffic in 128-bit ops; latency-bound problem
// with zero HBM traffic inside the 2248-step loop.

typedef __attribute__((ext_vector_type(16))) _Float16 v16h;
typedef __attribute__((ext_vector_type(8)))  _Float16 v8h;
typedef __attribute__((ext_vector_type(8)))  float    v8f;
typedef __attribute__((ext_vector_type(4)))  float    v4f;

#define BATCH    4
#define LSEQ     128
#define HDIM     64
#define VFHID    128
#define WDIM     5
#define ODIM     320     // W*H
#define NPAIR    10
#define LSLEN    16
#define LSD      16
#define NTHREADS 256

#define DEVINL __device__ __forceinline__

DEVINL float fast_tanh(float x) {
  // tanh(x) = 1 - 2/(exp(2x)+1); exp via v_exp_f32 (2^z), rcp via v_rcp_f32
  float e = __builtin_amdgcn_exp2f(x * 2.8853900817779268f);   // e^(2x)
  return 1.0f - 2.0f * __builtin_amdgcn_rcpf(e + 1.0f);
}

// One 16x16 output tile: D = A(16x32-chain) * B + C, C preloaded with bias
// (BIAS=true) or zero. A tiles pre-swizzled (512 halfs/tile, 16 contiguous
// per lane). B staged transposed in LDS: Bt[n][k], row stride KT*32 halfs.
template<int KT, bool BIAS, typename Post>
DEVINL void do_tile(int mt,
                    const _Float16* __restrict__ Asw,
                    const _Float16* __restrict__ BtIn,
                    const float* __restrict__ bias,
                    int lane, Post& post) {
  const int n  = lane & 15;
  const int hi = lane >> 4;
  const int mb = mt * 16 + hi * 8;
  v8f acc;
  if (BIAS) {
    v4f bA = *(const v4f*)(bias + mb);
    v4f bB = *(const v4f*)(bias + mb + 4);
#pragma unroll
    for (int r = 0; r < 4; ++r) { acc[r] = bA[r]; acc[r + 4] = bB[r]; }
  } else {
    acc = (v8f){0.f, 0.f, 0.f, 0.f, 0.f, 0.f, 0.f, 0.f};
  }
  const _Float16* aB = Asw + mt * (KT * 512) + lane * 16;
#pragma unroll
  for (int kt = 0; kt < KT; ++kt) {
    v16h a  = *(const v16h*)(aB + kt * 512);
    const _Float16* bp = BtIn + n * (KT * 32) + kt * 32 + hi * 8;
    v8h bl = *(const v8h*)bp, bh = *(const v8h*)(bp + 16);
    v16h bf = __builtin_shufflevector(bl, bh, 0, 1, 2, 3, 4, 5, 6, 7, 8, 9, 10, 11, 12, 13, 14, 15);
    acc = __builtin_amdgcn_wmma_f32_16x16x32_f16(false, a, false, bf,
                                                 (short)0, acc, false, false);
  }
  post(mt, acc, n, hi);
}

// MT = 8 -> one tile per wave; MT = 20 -> tiles wv, wv+8, (wv<4: wv+16).
template<int MT, int KT, bool BIAS, typename Post>
DEVINL void layer_pass(const _Float16* __restrict__ Asw,
                       const _Float16* __restrict__ BtIn,
                       const float* __restrict__ bias,
                       int wv, int lane, Post post) {
  do_tile<KT, BIAS>(wv, Asw, BtIn, bias, lane, post);
  if (MT > 8) {
    do_tile<KT, BIAS>(wv + 8, Asw, BtIn, bias, lane, post);
    if (wv < MT - 16)
      do_tile<KT, BIAS>(wv + 16, Asw, BtIn, bias, lane, post);
  }
}

__global__ __launch_bounds__(NTHREADS)
void logcde_kernel(const float* __restrict__ ts_g,
                   const float* __restrict__ logsig_g,
                   const float* __restrict__ x0_g,
                   const float* __restrict__ l1w_g, const float* __restrict__ l1b_g,
                   const float* __restrict__ l2w_g, const float* __restrict__ l2b_g,
                   const float* __restrict__ w0_g,  const float* __restrict__ b0_g,
                   const float* __restrict__ w1_g,  const float* __restrict__ b1_g,
                   const float* __restrict__ w2_g,  const float* __restrict__ b2_g,
                   const int*   __restrict__ pairs_g,
                   const int*   __restrict__ nsteps_g,
                   float* __restrict__ out_g) {
  // ---- LDS (~153 KB) ----
  __shared__ _Float16 Asw0[8 * 2 * 512];    // W0 128x64  f16, A-frag order
  __shared__ _Float16 Asw1[8 * 4 * 512];    // W1 128x128
  __shared__ _Float16 Asw2[20 * 4 * 512];   // W2 320x128
  __shared__ _Float16 BtY [16 * 64];        // layer-0 input, transposed [n][k]
  __shared__ _Float16 BtH0[16 * 128];       // layer-1 input
  __shared__ _Float16 BtH1[16 * 128];       // layer-2 input
  __shared__ float b0S[VFHID], b1S[VFHID], b2S[ODIM];
  __shared__ float g0S[VFHID], g1S[VFHID], g2S[ODIM];   // (1 - h^2) factors
  __shared__ float vfS[ODIM];                           // primal vf output
  __shared__ float DtT[WDIM * ODIM];                    // tangent outputs [j][m]
  __shared__ float tsS[LSEQ];
  __shared__ float lsS[LSLEN * LSD];
  __shared__ float yS[HDIM], k1S[HDIM], k2S[HDIM];
  __shared__ float logitsS[16];
  __shared__ int   iaS[NPAIR], ibS[NPAIR];

  const int tid  = (int)threadIdx.x;
  const int wv   = tid >> 5;      // wave id (wave32)
  const int lane = tid & 31;
  const int b    = (int)blockIdx.x;

  // ---------------- one-time init ----------------
  for (int i = tid; i < LSEQ; i += NTHREADS) tsS[i] = ts_g[i];
  for (int i = tid; i < LSLEN * LSD; i += NTHREADS)
    lsS[i] = logsig_g[b * (LSLEN * LSD) + i];
  for (int i = tid; i < VFHID; i += NTHREADS) { b0S[i] = b0_g[i]; b1S[i] = b1_g[i]; }
  for (int i = tid; i < ODIM;  i += NTHREADS) b2S[i] = b2_g[i];
  if (tid < NPAIR) { iaS[tid] = pairs_g[2 * tid] - 1; ibS[tid] = pairs_g[2 * tid + 1] - 1; }
  for (int i = tid; i < 16 * 64;  i += NTHREADS) BtY[i]  = (_Float16)0.0f;
  for (int i = tid; i < 16 * 128; i += NTHREADS) { BtH0[i] = (_Float16)0.0f; BtH1[i] = (_Float16)0.0f; }

  // pre-swizzle weights into 16x16x32 A-fragment order.
  // lane l (rows M = 16*mt + (l&15)); half h: j=h>>1, K = 16*(j>>2)+2*(j&3)+8*(l>>4)+(h&1)
  for (int idx = tid; idx < 8 * 2 * 512; idx += NTHREADS) {
    int tile = idx >> 9, w = idx & 511, ln = w >> 4, h = w & 15;
    int mt = tile >> 1, kt = tile & 1, j = h >> 1;
    int K = kt * 32 + 16 * (j >> 2) + 2 * (j & 3) + 8 * (ln >> 4) + (h & 1);
    int M = mt * 16 + (ln & 15);
    Asw0[idx] = (_Float16)w0_g[M * 64 + K];
  }
  for (int idx = tid; idx < 8 * 4 * 512; idx += NTHREADS) {
    int tile = idx >> 9, w = idx & 511, ln = w >> 4, h = w & 15;
    int mt = tile >> 2, kt = tile & 3, j = h >> 1;
    int K = kt * 32 + 16 * (j >> 2) + 2 * (j & 3) + 8 * (ln >> 4) + (h & 1);
    int M = mt * 16 + (ln & 15);
    Asw1[idx] = (_Float16)w1_g[M * 128 + K];
  }
  for (int idx = tid; idx < 20 * 4 * 512; idx += NTHREADS) {
    int tile = idx >> 9, w = idx & 511, ln = w >> 4, h = w & 15;
    int mt = tile >> 2, kt = tile & 3, j = h >> 1;
    int K = kt * 32 + 16 * (j >> 2) + 2 * (j & 3) + 8 * (ln >> 4) + (h & 1);
    int M = mt * 16 + (ln & 15);
    Asw2[idx] = (_Float16)w2_g[M * 128 + K];
  }

  // y0 = x0[b,1:] @ l1_w.T + l1_b ; stage into BtY column 0 (row n=0)
  if (tid < HDIM) {
    float a = l1b_g[tid];
#pragma unroll
    for (int w = 0; w < WDIM; ++w) a += x0_g[b * 6 + 1 + w] * l1w_g[tid * WDIM + w];
    yS[tid] = a;
    BtY[tid] = (_Float16)a;           // BtY[0*64 + k]
  }
  __syncthreads();

  const int   NST = nsteps_g[0];
  const float t0  = tsS[0];
  const float dtv = (tsS[LSEQ - 1] - t0) / (float)NST;

  // ---------------- field evaluation ----------------
  auto fieldEval = [&](float tcur, float* kOut) {
    // ---- primal pass: y -> h0 -> h1 -> vf_out (column 0) ----
    layer_pass<8, 2, true>(Asw0, BtY, b0S, wv, lane,
        [&](int mt, v8f acc, int n, int hi) {
      const int mb = mt * 16 + hi * 8;
      float v[8]; v8h o;
#pragma unroll
      for (int r = 0; r < 8; ++r) { v[r] = fast_tanh(acc[r]); o[r] = (_Float16)v[r]; }
      *(v8h*)(BtH0 + n * 128 + mb) = o;
      if (n == 0) {
        v4f gA, gB;
#pragma unroll
        for (int r = 0; r < 4; ++r) { gA[r] = 1.0f - v[r] * v[r]; gB[r] = 1.0f - v[r+4] * v[r+4]; }
        *(v4f*)(g0S + mb) = gA;  *(v4f*)(g0S + mb + 4) = gB;
      }
    });
    __syncthreads();
    layer_pass<8, 4, true>(Asw1, BtH0, b1S, wv, lane,
        [&](int mt, v8f acc, int n, int hi) {
      const int mb = mt * 16 + hi * 8;
      float v[8]; v8h o;
#pragma unroll
      for (int r = 0; r < 8; ++r) { v[r] = fast_tanh(acc[r]); o[r] = (_Float16)v[r]; }
      *(v8h*)(BtH1 + n * 128 + mb) = o;
      if (n == 0) {
        v4f gA, gB;
#pragma unroll
        for (int r = 0; r < 4; ++r) { gA[r] = 1.0f - v[r] * v[r]; gB[r] = 1.0f - v[r+4] * v[r+4]; }
        *(v4f*)(g1S + mb) = gA;  *(v4f*)(g1S + mb + 4) = gB;
      }
    });
    __syncthreads();
    layer_pass<20, 4, true>(Asw2, BtH1, b2S, wv, lane,
        [&](int mt, v8f acc, int n, int hi) {
      if (n == 0) {
        const int mb = mt * 16 + hi * 8;
        v8h o; v4f vA, vB, gA, gB;
#pragma unroll
        for (int r = 0; r < 8; ++r) {
          float v = fast_tanh(acc[r]);
          o[r] = (_Float16)v;
          if (r < 4) { vA[r] = v; gA[r] = 1.0f - v * v; }
          else       { vB[r-4] = v; gB[r-4] = 1.0f - v * v; }
        }
        *(v4f*)(vfS + mb) = vA;  *(v4f*)(vfS + mb + 4) = vB;
        *(v4f*)(g2S + mb) = gA;  *(v4f*)(g2S + mb + 4) = gB;
        // stage tangents: tangent j = vf_out row j -> BtY[j][k]
        *(v8h*)(BtY + (mb >> 6) * 64 + (mb & 63)) = o;
      }
    });
    __syncthreads();
    // ---- tangent pass: 5 JVP columns through linearized net ----
    layer_pass<8, 2, false>(Asw0, BtY, nullptr, wv, lane,
        [&](int mt, v8f acc, int n, int hi) {
      const int mb = mt * 16 + hi * 8;
      v4f gA = *(const v4f*)(g0S + mb), gB = *(const v4f*)(g0S + mb + 4);
      v8h o;
#pragma unroll
      for (int r = 0; r < 4; ++r) {
        o[r]   = (_Float16)(gA[r] * acc[r]);
        o[r+4] = (_Float16)(gB[r] * acc[r+4]);
      }
      *(v8h*)(BtH0 + n * 128 + mb) = o;
    });
    __syncthreads();
    layer_pass<8, 4, false>(Asw1, BtH0, nullptr, wv, lane,
        [&](int mt, v8f acc, int n, int hi) {
      const int mb = mt * 16 + hi * 8;
      v4f gA = *(const v4f*)(g1S + mb), gB = *(const v4f*)(g1S + mb + 4);
      v8h o;
#pragma unroll
      for (int r = 0; r < 4; ++r) {
        o[r]   = (_Float16)(gA[r] * acc[r]);
        o[r+4] = (_Float16)(gB[r] * acc[r+4]);
      }
      *(v8h*)(BtH1 + n * 128 + mb) = o;
    });
    __syncthreads();
    layer_pass<20, 4, false>(Asw2, BtH1, nullptr, wv, lane,
        [&](int mt, v8f acc, int n, int hi) {
      if (n < WDIM) {
        const int mb = mt * 16 + hi * 8;
        v4f gA = *(const v4f*)(g2S + mb), gB = *(const v4f*)(g2S + mb + 4);
        v4f dA, dB;
#pragma unroll
        for (int r = 0; r < 4; ++r) { dA[r] = gA[r] * acc[r]; dB[r] = gB[r] * acc[r+4]; }
        *(v4f*)(DtT + n * ODIM + mb)     = dA;
        *(v4f*)(DtT + n * ODIM + mb + 4) = dB;
      }
    });
    __syncthreads();
    // ---- combine: ls[1:6] @ vf_out + ls[6:16] @ lie ----
    if (tid < HDIM) {
      int lo = 0, hc = LSEQ;                       // searchsorted(ts, t), left
      while (lo < hc) { int mid = (lo + hc) >> 1; if (tsS[mid] < tcur) lo = mid + 1; else hc = mid; }
      int idx = lo >> 3; if (idx > LSLEN - 1) idx = LSLEN - 1;
      const float* ls = &lsS[idx * LSD];
      float a = 0.0f;
#pragma unroll
      for (int w = 0; w < WDIM; ++w) a += ls[1 + w] * vfS[w * HDIM + tid];
#pragma unroll
      for (int p = 0; p < NPAIR; ++p)
        a += ls[1 + WDIM + p] * (DtT[ibS[p] * ODIM + iaS[p] * HDIM + tid]
                               - DtT[iaS[p] * ODIM + ibS[p] * HDIM + tid]);
      kOut[tid] = a;
    }
  };

  // ---------------- Heun integration ----------------
  for (int i = 0; i < NST; ++i) {
    float tcur = t0 + (float)i * dtv;
    fieldEval(tcur, k1S);
    __syncthreads();
    if (tid < HDIM) {
      float y2 = yS[tid] + dtv * k1S[tid];
      BtY[tid] = (_Float16)y2;                    // stage y + dt*k1
    }
    __syncthreads();
    fieldEval(tcur + dtv, k2S);
    __syncthreads();
    if (tid < HDIM) {
      float yn = yS[tid] + 0.5f * dtv * (k1S[tid] + k2S[tid]);
      yS[tid] = yn;
      BtY[tid] = (_Float16)yn;                    // stage y for next step
    }
    __syncthreads();
  }

  // ---------------- readout: softmax(yT @ l2_w.T + l2_b) ----------------
  if (tid < 10) {
    float lg = l2b_g[tid];
    for (int h = 0; h < HDIM; ++h) lg += yS[h] * l2w_g[tid * HDIM + h];
    logitsS[tid] = lg;
  }
  __syncthreads();
  if (tid < 10) {
    float m = logitsS[0];
    for (int c = 1; c < 10; ++c) m = fmaxf(m, logitsS[c]);
    float s = 0.0f;
    for (int c = 0; c < 10; ++c)
      s += __builtin_amdgcn_exp2f((logitsS[c] - m) * 1.4426950408889634f);
    float e = __builtin_amdgcn_exp2f((logitsS[tid] - m) * 1.4426950408889634f);
    out_g[b * 10 + tid] = e / s;
  }
}

extern "C" void kernel_launch(void* const* d_in, const int* in_sizes, int n_in,
                              void* d_out, int out_size, void* d_ws, size_t ws_size,
                              hipStream_t stream) {
  (void)in_sizes; (void)n_in; (void)out_size; (void)d_ws; (void)ws_size;
  const float* ts    = (const float*)d_in[0];
  const float* logsg = (const float*)d_in[1];
  const float* x0    = (const float*)d_in[2];
  const float* l1w   = (const float*)d_in[3];
  const float* l1b   = (const float*)d_in[4];
  const float* l2w   = (const float*)d_in[5];
  const float* l2b   = (const float*)d_in[6];
  const float* w0    = (const float*)d_in[7];
  const float* b0    = (const float*)d_in[8];
  const float* w1    = (const float*)d_in[9];
  const float* b1    = (const float*)d_in[10];
  const float* w2    = (const float*)d_in[11];
  const float* b2    = (const float*)d_in[12];
  const int*   pairs = (const int*)d_in[13];
  const int*   nst   = (const int*)d_in[14];
  float* out = (float*)d_out;

  logcde_kernel<<<dim3(BATCH), dim3(NTHREADS), 0, stream>>>(
      ts, logsg, x0, l1w, l1b, l2w, l2b, w0, b0, w1, b1, w2, b2, pairs, nst, out);
}
